// GCNEstimator_11158325035469
// MI455X (gfx1250) — matrix operand order repaired and checked by reading, not compiled, not run on hardware
//
#include <hip/hip_runtime.h>
#include <hip/hip_bf16.h>
#include <hip/hip_fp16.h>

typedef __attribute__((ext_vector_type(16))) _Float16 v16h;
typedef __attribute__((ext_vector_type(8)))  _Float16 v8h;
typedef __attribute__((ext_vector_type(4)))  _Float16 v4h;
typedef __attribute__((ext_vector_type(8)))  float    v8f;

#define HIDDEN 64
#define DIN    130      // 128 emb + degree + seed mask
#define KPAD   160      // DIN padded to a multiple of 32 (5 WMMA K-tiles)
#define KT1    5        // K-tiles for layer 1
#define KT2    2        // K-tiles for layer 2

// Swizzled B-fragment storage: frag(kt,nt) -> 32 lanes x 16 contiguous halves.
// B_sw[(((kt*4)+nt)*32 + L)*16 + j] = W[(kt*32 + (L>=16?8:0) + (j&7) + (j>=8?16:0))*64 + nt*16 + (L&15)]
#define W1SW_HALVES (KT1 * 4 * 32 * 16)   // 10240 halves = 20 KB
#define W2SW_HALVES (KT2 * 4 * 32 * 16)   //  4096 halves =  8 KB

// ---------------------------------------------------------------------------
// CDNA5 async global->LDS copy (per-lane 16B), tracked on ASYNCcnt.
// ---------------------------------------------------------------------------
__device__ __forceinline__ void async_copy_b128(unsigned lds_off, unsigned long long gaddr) {
    asm volatile("global_load_async_to_lds_b128 %0, %1, off"
                 :: "v"(lds_off), "v"(gaddr) : "memory");
}
__device__ __forceinline__ void wait_async0() {
    asm volatile("s_wait_asynccnt 0x0" ::: "memory");
}

// ---------------------------------------------------------------------------
// Build f16 weight copies, pre-swizzled into WMMA B-fragment layout.
// ---------------------------------------------------------------------------
__global__ void prep_weights_kernel(const float* __restrict__ W1, const float* __restrict__ W2,
                                    _Float16* __restrict__ W1sw, _Float16* __restrict__ W2sw) {
    int tid    = blockIdx.x * blockDim.x + threadIdx.x;
    int stride = blockDim.x * gridDim.x;
    for (int idx = tid; idx < W1SW_HALVES; idx += stride) {
        int j  = idx & 15;
        int L  = (idx >> 4) & 31;
        int nt = (idx >> 9) & 3;
        int kt = idx >> 11;
        int k  = kt * 32 + ((L >> 4) << 3) + (j & 7) + ((j >> 3) << 4);
        int c  = nt * 16 + (L & 15);
        W1sw[idx] = (k < DIN) ? (_Float16)W1[k * HIDDEN + c] : (_Float16)0.0f;
    }
    for (int idx = tid; idx < W2SW_HALVES; idx += stride) {
        int j  = idx & 15;
        int L  = (idx >> 4) & 31;
        int nt = (idx >> 9) & 3;
        int kt = idx >> 11;
        int k  = kt * 32 + ((L >> 4) << 3) + (j & 7) + ((j >> 3) << 4);
        int c  = nt * 16 + (L & 15);
        W2sw[idx] = (_Float16)W2[k * HIDDEN + c];
    }
}

// ---------------------------------------------------------------------------
// GEMM1: out[N,64] = concat(emb, deg, mask)[N,160pad] @ W1[160,64] + b1
// One wave32 per 16-row strip, 4 col tiles. A fused concat w/ vector f32
// loads + packed cvt; B fragments are single v16h LDS loads (swizzled).
// ---------------------------------------------------------------------------
__global__ void __launch_bounds__(256)
gemm1_kernel(const float* __restrict__ emb, const float* __restrict__ deg,
             const float* __restrict__ msk, const _Float16* __restrict__ W1sw,
             const float* __restrict__ b1, float* __restrict__ out, int nRowTiles) {
    __shared__ __align__(32) _Float16 Wsh[W1SW_HALVES];   // 20 KB

    // Async-stage swizzled weights into LDS: 1280 chunks of 16B == 5 per thread.
    {
        unsigned ldsBase = (unsigned)(uintptr_t)&Wsh[0];
        const char* src  = (const char*)W1sw;
        for (int i = threadIdx.x; i < (W1SW_HALVES * 2) / 16; i += blockDim.x)
            async_copy_b128(ldsBase + i * 16, (unsigned long long)(src + i * 16));
        wait_async0();
        __syncthreads();
    }

    const int lane    = threadIdx.x & 31;
    const int wave    = threadIdx.x >> 5;
    const int rowTile = blockIdx.x * 8 + wave;
    if (rowTile >= nRowTiles) return;           // wave-uniform
    const int rowBase = rowTile * 16;

    const int r      = rowBase + (lane & 15);
    const int hiHalf = lane >> 4;               // 0 or 1
    const v16h* Wf   = reinterpret_cast<const v16h*>(Wsh);
    const float* rowp = emb + (size_t)r * 128;

    v8f c[4] = {};

    // K-tiles 0..3: all k < 128 -> pure vector loads.
    #pragma unroll
    for (int kt = 0; kt < 4; ++kt) {
        const int ka  = kt * 32 + hiHalf * 8;
        const float4 f0 = *reinterpret_cast<const float4*>(rowp + ka);
        const float4 f1 = *reinterpret_cast<const float4*>(rowp + ka + 4);
        const float4 f2 = *reinterpret_cast<const float4*>(rowp + ka + 16);
        const float4 f3 = *reinterpret_cast<const float4*>(rowp + ka + 20);
        v16h a;
        a[0]=(_Float16)f0.x; a[1]=(_Float16)f0.y; a[2]=(_Float16)f0.z; a[3]=(_Float16)f0.w;
        a[4]=(_Float16)f1.x; a[5]=(_Float16)f1.y; a[6]=(_Float16)f1.z; a[7]=(_Float16)f1.w;
        a[8]=(_Float16)f2.x; a[9]=(_Float16)f2.y; a[10]=(_Float16)f2.z; a[11]=(_Float16)f2.w;
        a[12]=(_Float16)f3.x; a[13]=(_Float16)f3.y; a[14]=(_Float16)f3.z; a[15]=(_Float16)f3.w;
        #pragma unroll
        for (int t = 0; t < 4; ++t) {
            v16h b = Wf[(kt * 4 + t) * 32 + lane];
            c[t] = __builtin_amdgcn_wmma_f32_16x16x32_f16(
                       false, a, false, b, (short)0, c[t], false, false);
        }
    }
    // K-tile 4 (k=128..159): only deg (k=128) and msk (k=129) in lanes 0-15;
    // everything else zero.  Branchless select keeps EXEC all-ones.
    {
        const float d = deg[r];
        const float m = msk[r];
        v16h a = {};
        a[0] = (_Float16)(hiHalf ? 0.0f : d);
        a[1] = (_Float16)(hiHalf ? 0.0f : m);
        #pragma unroll
        for (int t = 0; t < 4; ++t) {
            v16h b = Wf[(4 * 4 + t) * 32 + lane];
            c[t] = __builtin_amdgcn_wmma_f32_16x16x32_f16(
                       false, a, false, b, (short)0, c[t], false, false);
        }
    }

    const int m0 = hiHalf * 8;
    #pragma unroll
    for (int t = 0; t < 4; ++t) {
        const int cn   = t * 16 + (lane & 15);
        const float bb = b1[cn];
        #pragma unroll
        for (int j = 0; j < 8; ++j)
            out[(size_t)(rowBase + m0 + j) * HIDDEN + cn] = c[t][j] + bb;
    }
}

// ---------------------------------------------------------------------------
// GEMM2: out[N,64] = x_f16[N,64] @ W2[64,64] + b2
// ---------------------------------------------------------------------------
__global__ void __launch_bounds__(256)
gemm2_kernel(const _Float16* __restrict__ x, const _Float16* __restrict__ W2sw,
             const float* __restrict__ b2, float* __restrict__ out, int nRowTiles) {
    __shared__ __align__(32) _Float16 Wsh[W2SW_HALVES];   // 8 KB

    {
        unsigned ldsBase = (unsigned)(uintptr_t)&Wsh[0];
        const char* src  = (const char*)W2sw;
        for (int i = threadIdx.x; i < (W2SW_HALVES * 2) / 16; i += blockDim.x)
            async_copy_b128(ldsBase + i * 16, (unsigned long long)(src + i * 16));
        wait_async0();
        __syncthreads();
    }

    const int lane    = threadIdx.x & 31;
    const int wave    = threadIdx.x >> 5;
    const int rowTile = blockIdx.x * 8 + wave;
    if (rowTile >= nRowTiles) return;
    const int rowBase = rowTile * 16;

    const int r      = rowBase + (lane & 15);
    const int hiHalf = lane >> 4;
    const v16h* Wf   = reinterpret_cast<const v16h*>(Wsh);
    const _Float16* rp = x + (size_t)r * HIDDEN;

    v8f c[4] = {};

    #pragma unroll
    for (int kt = 0; kt < KT2; ++kt) {
        const int ka = kt * 32 + hiHalf * 8;
        v8h lo = *reinterpret_cast<const v8h*>(rp + ka);
        v8h hi = *reinterpret_cast<const v8h*>(rp + ka + 16);
        v16h a;
        #pragma unroll
        for (int i = 0; i < 8; ++i) { a[i] = lo[i]; a[i + 8] = hi[i]; }
        #pragma unroll
        for (int t = 0; t < 4; ++t) {
            v16h b = Wf[(kt * 4 + t) * 32 + lane];
            c[t] = __builtin_amdgcn_wmma_f32_16x16x32_f16(
                       false, a, false, b, (short)0, c[t], false, false);
        }
    }

    const int m0 = hiHalf * 8;
    #pragma unroll
    for (int t = 0; t < 4; ++t) {
        const int cn   = t * 16 + (lane & 15);
        const float bb = b2[cn];
        #pragma unroll
        for (int j = 0; j < 8; ++j)
            out[(size_t)(rowBase + m0 + j) * HIDDEN + cn] = c[t][j] + bb;
    }
}

// ---------------------------------------------------------------------------
// COO SpMM: acc[row[e], :] += vals[e] * x[col[e], :]  (wave per edge; feature
// matrix is L2-resident so gathers + f32 atomic scatters resolve in L2).
// ---------------------------------------------------------------------------
__global__ void __launch_bounds__(256)
spmm_kernel(const int* __restrict__ row, const int* __restrict__ col,
            const float* __restrict__ vals, const float* __restrict__ x,
            float* __restrict__ acc, int E) {
    const int e    = (blockIdx.x * blockDim.x + threadIdx.x) >> 5;
    const int lane = threadIdx.x & 31;
    if (e >= E) return;

    if (lane == 0 && e + 4096 < E) {
        __builtin_prefetch(row  + e + 4096, 0, 0);   // global_prefetch_b8
        __builtin_prefetch(col  + e + 4096, 0, 0);
        __builtin_prefetch(vals + e + 4096, 0, 0);
    }

    const int   rr = row[e];
    const int   cc = col[e];
    const float v  = vals[e];
    const float* xs = x   + (size_t)cc * HIDDEN;
    float*       as = acc + (size_t)rr * HIDDEN;

    atomicAdd(as + lane,      v * xs[lane]);
    atomicAdd(as + lane + 32, v * xs[lane + 32]);
}

// ---------------------------------------------------------------------------
// Fused ReLU + f32->f16 convert, 4 elements/thread.
// ---------------------------------------------------------------------------
__global__ void relu_cvt_kernel(const float4* __restrict__ in, v4h* __restrict__ out, int n4) {
    int i = blockIdx.x * blockDim.x + threadIdx.x;
    if (i < n4) {
        float4 f = in[i];
        v4h h;
        h[0] = (_Float16)fmaxf(f.x, 0.0f);
        h[1] = (_Float16)fmaxf(f.y, 0.0f);
        h[2] = (_Float16)fmaxf(f.z, 0.0f);
        h[3] = (_Float16)fmaxf(f.w, 0.0f);
        out[i] = h;
    }
}

// ---------------------------------------------------------------------------
// Fused ReLU + column sum over N rows -> gsum[64].
// ---------------------------------------------------------------------------
__global__ void __launch_bounds__(256)
colsum_kernel(const float* __restrict__ acc, float* __restrict__ gsum, int N) {
    __shared__ float ls[HIDDEN];
    const int t = threadIdx.x;
    if (t < HIDDEN) ls[t] = 0.0f;
    __syncthreads();

    const int f      = t & 63;
    const int rpb    = blockDim.x >> 6;
    const int r0     = blockIdx.x * rpb + (t >> 6);
    const int stride = gridDim.x * rpb;
    float s = 0.0f;
    for (int r = r0; r < N; r += stride)
        s += fmaxf(acc[(size_t)r * HIDDEN + f], 0.0f);
    atomicAdd(&ls[f], s);
    __syncthreads();
    if (t < HIDDEN) atomicAdd(&gsum[t], ls[t]);
}

// ---------------------------------------------------------------------------
// Final dot: out[0] = gsum . W_out + b_out
// ---------------------------------------------------------------------------
__global__ void finish_kernel(const float* __restrict__ gsum, const float* __restrict__ Wout,
                              const float* __restrict__ bout, float* __restrict__ out) {
    __shared__ float s[HIDDEN];
    const int t = threadIdx.x;
    s[t] = gsum[t] * Wout[t];
    __syncthreads();
    if (t == 0) {
        float a = bout[0];
        #pragma unroll
        for (int i = 0; i < HIDDEN; ++i) a += s[i];
        out[0] = a;
    }
}

// ---------------------------------------------------------------------------
extern "C" void kernel_launch(void* const* d_in, const int* in_sizes, int n_in,
                              void* d_out, int out_size, void* d_ws, size_t ws_size,
                              hipStream_t stream) {
    const float* emb   = (const float*)d_in[0];
    const float* deg   = (const float*)d_in[1];
    const float* msk   = (const float*)d_in[2];
    const int*   arow  = (const int*)  d_in[3];
    const int*   acol  = (const int*)  d_in[4];
    const float* avals = (const float*)d_in[5];
    const float* W1    = (const float*)d_in[6];
    const float* b1    = (const float*)d_in[7];
    const float* W2    = (const float*)d_in[8];
    const float* b2    = (const float*)d_in[9];
    const float* Wout  = (const float*)d_in[10];
    const float* bout  = (const float*)d_in[11];
    float* out = (float*)d_out;

    const int N = in_sizes[1];   // norm_degrees has N elements
    const int E = in_sizes[3];   // adj_row has E elements

    char* p = (char*)d_ws;
    auto carve = [&](size_t bytes) -> char* {
        char* q = p; p += (bytes + 255) & ~(size_t)255; return q;
    };
    float*    bufA = (float*)   carve((size_t)N * HIDDEN * sizeof(float));
    float*    bufB = (float*)   carve((size_t)N * HIDDEN * sizeof(float));
    _Float16* s1h  = (_Float16*)carve((size_t)N * HIDDEN * sizeof(_Float16));
    _Float16* W1sw = (_Float16*)carve(W1SW_HALVES * sizeof(_Float16));
    _Float16* W2sw = (_Float16*)carve(W2SW_HALVES * sizeof(_Float16));
    float*    gsum = (float*)   carve(HIDDEN * sizeof(float));

    prep_weights_kernel<<<32, 256, 0, stream>>>(W1, W2, W1sw, W2sw);

    const int nRowTiles = (N + 15) / 16;
    const int gemmBlks  = (nRowTiles + 7) / 8;
    const int spmmBlks  = (int)(((size_t)E * 32 + 255) / 256);
    const int n64       = N * HIDDEN;

    // Layer 1: GEMM -> SpMM(atomic) -> ReLU+cvt
    gemm1_kernel<<<gemmBlks, 256, 0, stream>>>(emb, deg, msk, W1sw, b1, bufA, nRowTiles);
    hipMemsetAsync(bufB, 0, (size_t)n64 * sizeof(float), stream);
    spmm_kernel<<<spmmBlks, 256, 0, stream>>>(arow, acol, avals, bufA, bufB, E);
    relu_cvt_kernel<<<(n64 / 4 + 255) / 256, 256, 0, stream>>>(
        (const float4*)bufB, (v4h*)s1h, n64 / 4);

    // Layer 2: GEMM -> SpMM(atomic) -> ReLU fused into column sum
    gemm2_kernel<<<gemmBlks, 256, 0, stream>>>(s1h, W2sw, b2, bufA, nRowTiles);
    hipMemsetAsync(bufB, 0, (size_t)n64 * sizeof(float), stream);
    spmm_kernel<<<spmmBlks, 256, 0, stream>>>(arow, acol, avals, bufA, bufB, E);

    hipMemsetAsync(gsum, 0, HIDDEN * sizeof(float), stream);
    colsum_kernel<<<1024, 256, 0, stream>>>(bufB, gsum, N);
    finish_kernel<<<1, HIDDEN, 0, stream>>>(gsum, Wout, bout, out);
}
